// ScRegNetGCNPretrainer_12386685681958
// MI455X (gfx1250) — matrix operand order repaired and checked by reading, not compiled, not run on hardware
//
#include <hip/hip_runtime.h>
#include <math.h>

typedef __bf16 bf16;
typedef __attribute__((ext_vector_type(16))) __bf16 v16bf;
typedef __attribute__((ext_vector_type(8)))  __bf16 v8bf;
typedef __attribute__((ext_vector_type(8)))  float  v8f;
typedef unsigned int v4u  __attribute__((ext_vector_type(4)));
typedef int          v8i  __attribute__((ext_vector_type(8)));
typedef int          v4i  __attribute__((ext_vector_type(4)));

#define NNODES 16384
#define DIN    1024
#define HID0   512
#define HID2   256
#define NPAIRS 100000
#define MPAD   100096            // NPAIRS padded to multiple of 128

#define LDS_STRIDE 40            // 32 k-elements + 8 bf16 pad per tile row

// Tensor Data Mover path for the bf16 B-tile staging.
// This toolchain exposes the 6-arg builtin:
//   (uint32x4 g0, int32x8 g1, int32x4 g2, int32x4 g3, int32x8 g4, i32 cpol)
#if defined(__has_builtin)
#  if __has_builtin(__builtin_amdgcn_tensor_load_to_lds) && \
      __has_builtin(__builtin_amdgcn_s_wait_tensorcnt)
#    define USE_TDM 1
#  endif
#endif
#ifndef USE_TDM
#  define USE_TDM 0
#endif

__device__ __forceinline__ float gelu_exact(float x) {
    return 0.5f * x * (1.0f + erff(x * 0.70710678118654752f));
}

// ---------------------------------------------------------------------------
// Generic bf16 WMMA GEMM:  C[M,N] = act( A[M,K] @ Bt[N,K]^T + bias )
//   AMODE: 0 = A fp32 row-major (cvt->bf16 while staging)
//          1 = A bf16 row-major
//          2 = A gathered pair rows from fp32 EMB[*,256] via idx0/idx1
//   ACT  : 0 none, 1 relu, 2 exact gelu
//   TOUT : write C transposed ([N][M], bf16, packed 16B stores)  -> next B
//   FOUT : write C fp32 row-major ; else bf16 row-major
// Block: 256 threads = 8 waves (2 x 4). Block tile 128x128, BK=32.
// Each wave: 64x32 output = 4x2 v_wmma_f32_16x16x32_bf16 accumulators.
// A tile: register double-buffered. B tile: TDM (tensor_load_to_lds) with
// LDS padding via D# pad fields, or register staging fallback.
// ---------------------------------------------------------------------------
template<int AMODE, int BIAS, int ACT, int TOUT, int FOUT>
__global__ __launch_bounds__(256)
void gemm_wmma_bf16(const float* __restrict__ Af, const bf16* __restrict__ Ab,
                    const bf16* __restrict__ Bt, const float* __restrict__ bias,
                    bf16* __restrict__ Cb, float* __restrict__ Cf,
                    int M, int N, int K, int lda,
                    const int* __restrict__ idx0, const int* __restrict__ idx1,
                    int P)
{
    __shared__ bf16 As[128 * LDS_STRIDE];
    __shared__ bf16 Bs[128 * LDS_STRIDE];

    const int bm = blockIdx.y, bn = blockIdx.x;
    const int tid  = threadIdx.x;
    const int r    = tid >> 1;     // 0..127 : tile row to stage
    const int h    = tid & 1;      // which 16-element half of BK=32
    const int wave = tid >> 5;
    const int lane = tid & 31;
    const int wm   = wave >> 2;    // 0..1  : 64-row strip
    const int wn   = wave & 3;     // 0..3  : 32-col strip
    const int l15  = lane & 15;
    const int lh   = lane >> 4;

    const int nk = K >> 5;

    // ---- A tile -> registers (16 bf16 per thread) ----
    auto loadA = [&](int kk, v8bf& lo, v8bf& hi) {
        const int k0 = kk << 5;
        if (AMODE == 0) {                      // fp32 A (x / adj), cvt->bf16
            const float* src = Af + (size_t)(bm * 128 + r) * lda + k0 + h * 16;
            if (kk + 1 < nk) __builtin_prefetch(src + 32, 0, 0);
            float4 f0 = *(const float4*)(src + 0);
            float4 f1 = *(const float4*)(src + 4);
            float4 f2 = *(const float4*)(src + 8);
            float4 f3 = *(const float4*)(src + 12);
            lo[0]=(bf16)f0.x; lo[1]=(bf16)f0.y; lo[2]=(bf16)f0.z; lo[3]=(bf16)f0.w;
            lo[4]=(bf16)f1.x; lo[5]=(bf16)f1.y; lo[6]=(bf16)f1.z; lo[7]=(bf16)f1.w;
            hi[0]=(bf16)f2.x; hi[1]=(bf16)f2.y; hi[2]=(bf16)f2.z; hi[3]=(bf16)f2.w;
            hi[4]=(bf16)f3.x; hi[5]=(bf16)f3.y; hi[6]=(bf16)f3.z; hi[7]=(bf16)f3.w;
        } else if (AMODE == 1) {               // bf16 A (hidden activations)
            const bf16* src = Ab + (size_t)(bm * 128 + r) * lda + k0 + h * 16;
            if (kk + 1 < nk) __builtin_prefetch(src + 32, 0, 0);
            lo = *(const v8bf*)(src);
            hi = *(const v8bf*)(src + 8);
        } else {                               // gathered pair rows from EMB
            int p  = bm * 128 + r;
            int pc = (p < P) ? p : 0;          // clamp pad rows (outputs unused)
            int kg = k0 + h * 16;              // BK block entirely on one side
            int rowi = (kg < HID2) ? idx0[pc] : idx1[pc];
            const float* src = Af + (size_t)rowi * HID2 + (kg & (HID2 - 1));
            float4 f0 = *(const float4*)(src + 0);
            float4 f1 = *(const float4*)(src + 4);
            float4 f2 = *(const float4*)(src + 8);
            float4 f3 = *(const float4*)(src + 12);
            lo[0]=(bf16)f0.x; lo[1]=(bf16)f0.y; lo[2]=(bf16)f0.z; lo[3]=(bf16)f0.w;
            lo[4]=(bf16)f1.x; lo[5]=(bf16)f1.y; lo[6]=(bf16)f1.z; lo[7]=(bf16)f1.w;
            hi[0]=(bf16)f2.x; hi[1]=(bf16)f2.y; hi[2]=(bf16)f2.z; hi[3]=(bf16)f2.w;
            hi[4]=(bf16)f3.x; hi[5]=(bf16)f3.y; hi[6]=(bf16)f3.z; hi[7]=(bf16)f3.w;
        }
    };

#if USE_TDM
    // One TDM descriptor moves the whole 128x32 bf16 B tile into LDS.
    // D# pad fields (interval=16 DWORDs, amount=4 DWORDs) recreate the
    // 40-element LDS row stride used by the fragment readers.
    auto tdmIssueB = [&](int kk) {
        const int k0 = kk << 5;
        const bf16* gsrc = Bt + (size_t)(bn * 128) * K + k0;
        unsigned long long ga = (unsigned long long)(size_t)gsrc;
        unsigned int ldsa = (unsigned int)(size_t)(&Bs[0]);
        v4u g0 = { 1u,                                   // count=1, user mode
                   ldsa,                                 // lds_addr
                   (unsigned int)(ga & 0xFFFFFFFFu),
                   (unsigned int)((ga >> 32) & 0x01FFFFFFu) | (2u << 30) }; // type=2
        unsigned int dim0 = (unsigned int)K;             // tensor_dim0 (elements)
        unsigned int dim1 = (unsigned int)N;             // tensor_dim1 (rows)
        unsigned long long st0 = (unsigned long long)K;  // dim0 stride (elements)
        v8i g1 = {
            (int)((1u << 16) | (1u << 20) | (3u << 22) | (3u << 25)), // bf16, pad on
            (int)((dim0 & 0xFFFFu) << 16),                // atomic addr 0 | dim0.lo
            (int)(((dim0 >> 16) & 0xFFFFu) | ((dim1 & 0xFFFFu) << 16)),
            (int)(((dim1 >> 16) & 0xFFFFu) | (32u << 16)),// tile_dim0 = 32
            (int)(128u),                                  // tile_dim1 = 128
            (int)(unsigned int)(st0 & 0xFFFFFFFFu),
            (int)(unsigned int)((st0 >> 32) & 0xFFFFu),   // stride1 = 0
            0 };
        v4i z4 = { 0, 0, 0, 0 };
        v8i z8 = { 0, 0, 0, 0, 0, 0, 0, 0 };
        __builtin_amdgcn_tensor_load_to_lds(g0, g1, z4, z4, z8, 0);
    };
#else
    auto loadB = [&](int kk, v8bf& lo, v8bf& hi) {
        const int k0 = kk << 5;
        const bf16* src = Bt + (size_t)(bn * 128 + r) * K + k0 + h * 16;
        lo = *(const v8bf*)(src);
        hi = *(const v8bf*)(src + 8);
    };
#endif

    v8f acc[4][2];
    #pragma unroll
    for (int i = 0; i < 4; ++i)
        #pragma unroll
        for (int j = 0; j < 2; ++j)
            #pragma unroll
            for (int e = 0; e < 8; ++e) acc[i][j][e] = 0.0f;

    v8bf raLo, raHi;
    loadA(0, raLo, raHi);
#if USE_TDM
    if (wave == 0) tdmIssueB(0);
#else
    v8bf rbLo, rbHi;
    loadB(0, rbLo, rbHi);
#endif

    for (int kk = 0; kk < nk; ++kk) {
        // ---- commit staged tiles to LDS ----
        {
            bf16* dst = &As[r * LDS_STRIDE + h * 16];
            *(v8bf*)dst = raLo;
            *(v8bf*)(dst + 8) = raHi;
        }
#if USE_TDM
        if (wave == 0) __builtin_amdgcn_s_wait_tensorcnt(0);
#else
        {
            bf16* dst = &Bs[r * LDS_STRIDE + h * 16];
            *(v8bf*)dst = rbLo;
            *(v8bf*)(dst + 8) = rbHi;
        }
#endif
        __syncthreads();

        // ---- prefetch next A tile into registers (overlaps WMMAs) ----
        if (kk + 1 < nk) loadA(kk + 1, raLo, raHi);

        // ---- build fragments per ISA 16-bit A/B layouts, issue WMMAs ----
        v16bf afrag[4], bfrag[2];
        #pragma unroll
        for (int mt = 0; mt < 4; ++mt) {
            // A 16x32: lanes 0-15/16-31 both M=0..15; half selects K {0..7,16..23}/{8..15,24..31}
            const int row = wm * 64 + mt * 16 + l15;
            v8bf lo = *(const v8bf*)&As[row * LDS_STRIDE + lh * 8];
            v8bf hi = *(const v8bf*)&As[row * LDS_STRIDE + 16 + lh * 8];
            #pragma unroll
            for (int i = 0; i < 8; ++i) { afrag[mt][i] = lo[i]; afrag[mt][i + 8] = hi[i]; }
        }
        #pragma unroll
        for (int nt = 0; nt < 2; ++nt) {
            // B 32x16: lane = N column; half selects K {0..7,16..23}/{8..15,24..31}
            const int col = wn * 32 + nt * 16 + l15;
            v8bf lo = *(const v8bf*)&Bs[col * LDS_STRIDE + lh * 8];
            v8bf hi = *(const v8bf*)&Bs[col * LDS_STRIDE + 16 + lh * 8];
            #pragma unroll
            for (int i = 0; i < 8; ++i) { bfrag[nt][i] = lo[i]; bfrag[nt][i + 8] = hi[i]; }
        }
        #pragma unroll
        for (int mt = 0; mt < 4; ++mt)
            #pragma unroll
            for (int nt = 0; nt < 2; ++nt)
                acc[mt][nt] = __builtin_amdgcn_wmma_f32_16x16x32_bf16(
                    false, afrag[mt], false, bfrag[nt],
                    (short)0, acc[mt][nt], false, false);
        __syncthreads();

        // ---- kick off next B tile (DMA overlaps next A stage + WMMAs) ----
        if (kk + 1 < nk) {
#if USE_TDM
            if (wave == 0) tdmIssueB(kk + 1);
#else
            loadB(kk + 1, rbLo, rbHi);
#endif
        }
    }

    // ---- epilogue: bias + activation + store ----
    // C layout: lane = N col (l15); VGPR e = row (lanes 16-31 offset +8)
    #pragma unroll
    for (int mt = 0; mt < 4; ++mt) {
        #pragma unroll
        for (int nt = 0; nt < 2; ++nt) {
            const int col  = bn * 128 + wn * 32 + nt * 16 + l15;
            const int row0 = bm * 128 + wm * 64 + mt * 16 + lh * 8;
            float vals[8];
            #pragma unroll
            for (int e = 0; e < 8; ++e) {
                float v = acc[mt][nt][e];
                if constexpr (BIAS) v += bias[col];
                if constexpr (ACT == 1) v = fmaxf(v, 0.0f);
                if constexpr (ACT == 2) v = gelu_exact(v);
                vals[e] = v;
            }
            if constexpr (TOUT) {     // transposed bf16 [N][M]: 8 rows contiguous
                v8bf pk;
                #pragma unroll
                for (int e = 0; e < 8; ++e) pk[e] = (bf16)vals[e];
                *(v8bf*)(Cb + (size_t)col * M + row0) = pk;
            } else if constexpr (FOUT) {  // fp32 row-major
                #pragma unroll
                for (int e = 0; e < 8; ++e)
                    Cf[(size_t)(row0 + e) * N + col] = vals[e];
            } else {                  // bf16 row-major
                #pragma unroll
                for (int e = 0; e < 8; ++e)
                    Cb[(size_t)(row0 + e) * N + col] = (bf16)vals[e];
            }
        }
    }
}

// fp32 W[K][N] -> bf16 Wt[N][K]  (one-time weight transpose/convert)
__global__ __launch_bounds__(256)
void transpose_f32_to_bf16(const float* __restrict__ W, bf16* __restrict__ Wt,
                           int K, int N)
{
    int i = blockIdx.x * 256 + threadIdx.x;
    if (i >= K * N) return;
    int k = i / N, n = i - k * N;
    Wt[(size_t)n * K + k] = (bf16)W[i];
}

// logits[P,3] = hdn(bf16)[P,256] @ Wh2[256,3] + bh2   (tiny N -> plain VALU)
__global__ __launch_bounds__(256)
void head_logits(const bf16* __restrict__ hdn, const float* __restrict__ Wh2,
                 const float* __restrict__ bh2, float* __restrict__ out, int P)
{
    int p = blockIdx.x * 256 + threadIdx.x;
    if (p >= P) return;
    float a0 = bh2[0], a1 = bh2[1], a2 = bh2[2];
    const bf16* hrow = hdn + (size_t)p * HID2;
    #pragma unroll 4
    for (int k = 0; k < HID2; ++k) {
        float v = (float)hrow[k];
        a0 += v * Wh2[k * 3 + 0];
        a1 += v * Wh2[k * 3 + 1];
        a2 += v * Wh2[k * 3 + 2];
    }
    out[(size_t)p * 3 + 0] = a0;
    out[(size_t)p * 3 + 1] = a1;
    out[(size_t)p * 3 + 2] = a2;
}

extern "C" void kernel_launch(void* const* d_in, const int* in_sizes, int n_in,
                              void* d_out, int out_size, void* d_ws, size_t ws_size,
                              hipStream_t stream)
{
    const float* x   = (const float*)d_in[0];
    const float* adj = (const float*)d_in[1];
    const int*   tf  = (const int*)d_in[2];
    const int*   gi  = (const int*)d_in[3];
    const float* W0  = (const float*)d_in[4];
    const float* b0  = (const float*)d_in[5];
    const float* W1  = (const float*)d_in[6];
    const float* b1  = (const float*)d_in[7];
    const float* W2  = (const float*)d_in[8];
    const float* b2  = (const float*)d_in[9];
    const float* Wh1 = (const float*)d_in[10];
    const float* bh1 = (const float*)d_in[11];
    const float* Wh2 = (const float*)d_in[12];
    const float* bh2 = (const float*)d_in[13];
    float* logits = (float*)d_out;

    // ---- workspace layout (bytes) ----
    char* ws = (char*)d_ws;
    size_t off = 0;
    auto alloc = [&](size_t bytes) { char* p = ws + off; off += (bytes + 255) & ~(size_t)255; return p; };
    bf16*  W0t  = (bf16*)alloc((size_t)HID0 * DIN  * 2);     // [512][1024]
    bf16*  W1t  = (bf16*)alloc((size_t)HID0 * HID0 * 2);     // [512][512]
    bf16*  W2t  = (bf16*)alloc((size_t)HID2 * HID0 * 2);     // [256][512]
    bf16*  Wh1t = (bf16*)alloc((size_t)HID2 * (2*HID2) * 2); // [256][512]
    bf16*  Tbuf = (bf16*)alloc((size_t)HID0 * NNODES * 2);   // K-major activations
    bf16*  Hbuf = (bf16*)alloc((size_t)NNODES * HID0 * 2);   // row-major activations
    float* EMB  = (float*)alloc((size_t)NNODES * HID2 * 4);
    bf16*  hdn  = (bf16*)alloc((size_t)MPAD * HID2 * 2);

    // ---- weight transposes/converts (re-run every call: deterministic) ----
    transpose_f32_to_bf16<<<(DIN*HID0 + 255)/256, 256, 0, stream>>>(W0,  W0t,  DIN,    HID0);
    transpose_f32_to_bf16<<<(HID0*HID0 + 255)/256, 256, 0, stream>>>(W1,  W1t,  HID0,   HID0);
    transpose_f32_to_bf16<<<(HID0*HID2 + 255)/256, 256, 0, stream>>>(W2,  W2t,  HID0,   HID2);
    transpose_f32_to_bf16<<<(2*HID2*HID2 + 255)/256, 256, 0, stream>>>(Wh1, Wh1t, 2*HID2, HID2);

    dim3 blk(256);

    // 1) T0^T = (x @ W0)^T                [512][16384] bf16
    gemm_wmma_bf16<0,0,0,1,0><<<dim3(HID0/128, NNODES/128), blk, 0, stream>>>(
        x, nullptr, W0t, nullptr, Tbuf, nullptr,
        NNODES, HID0, DIN, DIN, nullptr, nullptr, 0);
    // 2) H0 = relu(adj @ T0 + b0)         [16384][512] bf16
    gemm_wmma_bf16<0,1,1,0,0><<<dim3(HID0/128, NNODES/128), blk, 0, stream>>>(
        adj, nullptr, Tbuf, b0, Hbuf, nullptr,
        NNODES, HID0, NNODES, NNODES, nullptr, nullptr, 0);
    // 3) T1^T = (H0 @ W1)^T               [512][16384] bf16
    gemm_wmma_bf16<1,0,0,1,0><<<dim3(HID0/128, NNODES/128), blk, 0, stream>>>(
        nullptr, Hbuf, W1t, nullptr, Tbuf, nullptr,
        NNODES, HID0, HID0, HID0, nullptr, nullptr, 0);
    // 4) H1 = relu(adj @ T1 + b1)         [16384][512] bf16
    gemm_wmma_bf16<0,1,1,0,0><<<dim3(HID0/128, NNODES/128), blk, 0, stream>>>(
        adj, nullptr, Tbuf, b1, Hbuf, nullptr,
        NNODES, HID0, NNODES, NNODES, nullptr, nullptr, 0);
    // 5) T2^T = (H1 @ W2)^T               [256][16384] bf16
    gemm_wmma_bf16<1,0,0,1,0><<<dim3(HID2/128, NNODES/128), blk, 0, stream>>>(
        nullptr, Hbuf, W2t, nullptr, Tbuf, nullptr,
        NNODES, HID2, HID0, HID0, nullptr, nullptr, 0);
    // 6) EMB = adj @ T2 + b2              [16384][256] fp32
    gemm_wmma_bf16<0,1,0,0,1><<<dim3(HID2/128, NNODES/128), blk, 0, stream>>>(
        adj, nullptr, Tbuf, b2, nullptr, EMB,
        NNODES, HID2, NNODES, NNODES, nullptr, nullptr, 0);
    // 7) hdn = gelu([emb[tf] | emb[gene]] @ Wh1 + bh1)   [MPAD][256] bf16
    gemm_wmma_bf16<2,1,2,0,0><<<dim3(HID2/128, MPAD/128), blk, 0, stream>>>(
        EMB, nullptr, Wh1t, bh1, hdn, nullptr,
        MPAD, HID2, 2*HID2, HID2, tf, gi, NPAIRS);
    // 8) logits = hdn @ Wh2 + bh2         [100000][3] fp32
    head_logits<<<(NPAIRS + 255)/256, 256, 0, stream>>>(hdn, Wh2, bh2, logits, NPAIRS);
}
